// ConvAttention_61083024883911
// MI455X (gfx1250) — matrix operand order repaired and checked by reading, not compiled, not run on hardware
//
#include <hip/hip_runtime.h>
#include <hip/hip_bf16.h>

// ---------------------------------------------------------------------------
// ConvAttention (ViT-style) for MI455X / gfx1250, wave32, WMMA f16->f32.
// B=64, N=197, D=768, HEADS=12, DH=64, INNER=768, HW=14
//
// GEMM: 256x64 tile, K-step 32, 8 waves; each wave owns 32 rows x 64 cols
// (8 wmma/K-step from 2 A-frags + 4 B-frags, ds_load_2addr_b64 gathers).
// Double-buffered LDS + register prefetch; steady-state loop is branch-free
// (last K-step peeled, unroll-2 makes buffer parity static). Weights are
// pre-converted per launch to f16 [n][k]; A lives in workspace padded to a
// 256-row multiple so A loads are unguarded float4. All elementwise passes
// are float4-vectorized (memory-bound at 23.3 TB/s HBM).
// ---------------------------------------------------------------------------

typedef __attribute__((ext_vector_type(16))) _Float16 v16h;
typedef __attribute__((ext_vector_type(2)))  _Float16 h2;
typedef __attribute__((ext_vector_type(8)))  float    v8f;

#define BQ   64
#define NTOK 197
#define DIM  768
#define NIMG 196
#define QKVW 2304

// ------------------------------ GEMM ---------------------------------------
#define BM 256
#define BN 64
#define BK 32
#define PITCH 18                 // LDS row pitch in dwords (36 halves, 72 B)
#define A_BUF (BM * PITCH)       // 4608 dwords per A buffer
#define B_BUF (BN * PITCH)       // 1152 dwords per B buffer
#define SB_OFF (2 * A_BUF)

__device__ __forceinline__ unsigned pack2h(float a, float b)
{
    h2 p;
    p[0] = (_Float16)a;
    p[1] = (_Float16)b;
    return __builtin_bit_cast(unsigned, p);
}

__device__ __forceinline__ void load_a_tile(const float* __restrict__ A, int lda,
                                            int m0, int k0, int tid, float4* regs)
{
    const int seg   = tid & 7;        // 8 float4 segments per 32-float row
    const int rbase = tid >> 3;       // 32 rows per i-step
    #pragma unroll
    for (int i = 0; i < 8; ++i) {
        const int r = i * 32 + rbase;
        regs[i] = *(const float4*)(A + (size_t)(m0 + r) * lda + k0 + seg * 4);
    }
}

__device__ __forceinline__ void store_a_tile(unsigned* sA, int tid, const float4* regs)
{
    const int seg   = tid & 7;
    const int rbase = tid >> 3;
    #pragma unroll
    for (int i = 0; i < 8; ++i) {
        const int r = i * 32 + rbase;
        const float4 v = regs[i];
        *(uint2*)(sA + r * PITCH + seg * 2) = make_uint2(pack2h(v.x, v.y), pack2h(v.z, v.w));
    }
}

__device__ __forceinline__ void load_b_tile(const _Float16* __restrict__ Wh, int K,
                                            int n0, int k0, int tid, uint4* reg)
{
    const int r     = tid >> 2;       // 64 rows
    const int chunk = tid & 3;        // 4 x 8 halves per 32-half row
    *reg = *(const uint4*)(Wh + (size_t)(n0 + r) * K + k0 + chunk * 8);
}

__device__ __forceinline__ void store_b_tile(unsigned* sB, int tid, const uint4* reg)
{
    const int r     = tid >> 2;
    const int chunk = tid & 3;
    *(uint2*)(sB + r * PITCH + chunk * 4)     = make_uint2(reg->x, reg->y);
    *(uint2*)(sB + r * PITCH + chunk * 4 + 2) = make_uint2(reg->z, reg->w);
}

__device__ __forceinline__ void compute_step(const unsigned* lds, int cur,
                                             int wave, int lo, int hi, v8f acc[2][4])
{
    const _Float16* hA = (const _Float16*)(lds + cur * A_BUF);
    const _Float16* hB = (const _Float16*)(lds + SB_OFF + cur * B_BUF);

    // A fragments (ISA 16-bit A 16x32 layout: K = (v&3)*2 + (v>>2)*16 + hi*8)
    v16h af[2];
    #pragma unroll
    for (int mt = 0; mt < 2; ++mt) {
        const int abase = (wave * 32 + mt * 16 + lo) * (2 * PITCH);
        #pragma unroll
        for (int vv = 0; vv < 8; ++vv) {
            const int kk = ((vv & 3) * 2) + ((vv >> 2) * 16) + hi * 8;
            af[mt][2 * vv]     = hA[abase + kk];
            af[mt][2 * vv + 1] = hA[abase + kk + 1];
        }
    }
    #pragma unroll
    for (int nt = 0; nt < 4; ++nt) {
        // B fragment: lane owns column n = lo; K = (v&3)*2 + (v>>2)*16 + hi*8
        v16h bf;
        const int bbase = (nt * 16 + lo) * (2 * PITCH);
        #pragma unroll
        for (int vv = 0; vv < 8; ++vv) {
            const int kk = ((vv & 3) * 2) + ((vv >> 2) * 16) + hi * 8;
            bf[2 * vv]     = hB[bbase + kk];
            bf[2 * vv + 1] = hB[bbase + kk + 1];
        }
        acc[0][nt] = __builtin_amdgcn_wmma_f32_16x16x32_f16(
            false, af[0], false, bf, (short)0, acc[0][nt], false, false);
        acc[1][nt] = __builtin_amdgcn_wmma_f32_16x16x32_f16(
            false, af[1], false, bf, (short)0, acc[1][nt], false, false);
    }
}

__global__ __launch_bounds__(256)
void gemm_wmma_f16(const float* __restrict__ A, int lda, int M, int K,
                   const _Float16* __restrict__ Wh,                 // [n][k], ld = K
                   const float* __restrict__ bias,                  // nullable
                   const float* __restrict__ resid, int ldr,        // nullable
                   float* __restrict__ Out, int ldo,
                   int rowmap)                                      // 0: identity, 1: img tokens
{
    __shared__ unsigned lds[2 * A_BUF + 2 * B_BUF];   // 46080 B

    const int tid  = threadIdx.x;
    const int wave = tid >> 5;
    const int lane = tid & 31;
    const int m0   = blockIdx.x * BM;
    const int n0   = blockIdx.y * BN;
    const int lo   = lane & 15;
    const int hi   = (lane >> 4) & 1;
    const int KT   = K / BK;          // 24 for all GEMMs here

    v8f acc[2][4] = {};
    float4 aR[8];
    uint4  bR;

    // prologue: stage K-step 0 into buffer 0
    load_a_tile(A, lda, m0, 0, tid, aR);
    load_b_tile(Wh, K, n0, 0, tid, &bR);
    store_a_tile(lds, tid, aR);
    store_b_tile(lds + SB_OFF, tid, &bR);
    __syncthreads();

    // steady state: branch-free (last step peeled); unroll 2 -> static parity
    #pragma unroll 2
    for (int kt = 0; kt < KT - 1; ++kt) {
        load_a_tile(A, lda, m0, (kt + 1) * BK, tid, aR);   // overlap with wmma
        load_b_tile(Wh, K, n0, (kt + 1) * BK, tid, &bR);
        compute_step(lds, kt & 1, wave, lo, hi, acc);
        __syncthreads();
        store_a_tile(lds + ((kt + 1) & 1) * A_BUF, tid, aR);
        store_b_tile(lds + SB_OFF + ((kt + 1) & 1) * B_BUF, tid, &bR);
        __syncthreads();
    }
    compute_step(lds, (KT - 1) & 1, wave, lo, hi, acc);

    // ---- epilogue: C/D layout: N = lo, M = r + 8*hi ----
    #pragma unroll
    for (int mt = 0; mt < 2; ++mt) {
        #pragma unroll
        for (int nt = 0; nt < 4; ++nt) {
            const int gc = n0 + nt * 16 + lo;
            const float bv = bias ? bias[gc] : 0.0f;
            #pragma unroll
            for (int r = 0; r < 8; ++r) {
                const int grow = m0 + wave * 32 + mt * 16 + r + 8 * hi;
                if (grow < M) {
                    size_t orow = (size_t)grow;
                    if (rowmap == 1) {
                        const int b  = grow / NIMG;
                        const int hw = grow - b * NIMG;
                        orow = (size_t)b * NTOK + 1 + hw;
                    }
                    float val = acc[mt][nt][r] + bv;
                    if (resid) val += resid[orow * (size_t)ldr + gc];
                    Out[orow * (size_t)ldo + gc] = val;
                }
            }
        }
    }
}

// ------------------- weight prep: f32 -> f16 -------------------------------
__global__ __launch_bounds__(256)
void convert_f16(const float* __restrict__ W, _Float16* __restrict__ Wh, int n4)
{
    const int idx = blockIdx.x * 256 + threadIdx.x;   // one float4 per thread
    if (idx >= n4) return;
    const float4 v = ((const float4*)W)[idx];
    ((uint2*)Wh)[idx] = make_uint2(pack2h(v.x, v.y), pack2h(v.z, v.w));
}

// W is [K][N] f32; Wh is [N][K] f16. K, N multiples of 32.
__global__ __launch_bounds__(256)
void transpose_f16(const float* __restrict__ W, int K, int N, _Float16* __restrict__ Wh)
{
    __shared__ float tile[32][33];
    const int k0 = blockIdx.x * 32;
    const int n0 = blockIdx.y * 32;
    const int tx = threadIdx.x & 31;
    const int ty = threadIdx.x >> 5;   // 0..7
    #pragma unroll
    for (int i = 0; i < 32; i += 8)
        tile[ty + i][tx] = W[(size_t)(k0 + ty + i) * N + n0 + tx];
    __syncthreads();
    #pragma unroll
    for (int i = 0; i < 32; i += 8)
        Wh[(size_t)(n0 + ty + i) * K + k0 + tx] = (_Float16)tile[tx][ty + i];
}

// ------------------ depthwise 3x3 SAME (float4 channels) -------------------
__global__ __launch_bounds__(256)
void dwconv3x3(const float* __restrict__ in, int istride, int ioff,
               const float* __restrict__ w, const float* __restrict__ bia,
               float* __restrict__ out)
{
    const int C4 = DIM / 4;                                // 192
    const size_t idx   = (size_t)blockIdx.x * 256 + threadIdx.x;
    const size_t total = (size_t)BQ * NIMG * C4;
    if (idx >= total) return;
    const int c4 = (int)(idx % C4);
    const int t  = (int)(idx / C4);
    const int hw = t % NIMG;
    const int b  = t / NIMG;
    const int y  = hw / 14;
    const int x  = hw % 14;
    const int c  = c4 * 4;

    float4 s = *(const float4*)(bia + c);
    #pragma unroll
    for (int dy = 0; dy < 3; ++dy) {
        const int yy = y + dy - 1;
        if (yy < 0 || yy > 13) continue;
        #pragma unroll
        for (int dx = 0; dx < 3; ++dx) {
            const int xx = x + dx - 1;
            if (xx < 0 || xx > 13) continue;
            const size_t row = (size_t)b * NTOK + 1 + (yy * 14 + xx);
            const float4 iv = *(const float4*)(in + row * istride + ioff + c);
            const int tap = dy * 3 + dx;
            s.x += iv.x * w[(c + 0) * 9 + tap];
            s.y += iv.y * w[(c + 1) * 9 + tap];
            s.z += iv.z * w[(c + 2) * 9 + tap];
            s.w += iv.w * w[(c + 3) * 9 + tap];
        }
    }
    *(float4*)(out + (size_t)(b * NIMG + hw) * DIM + c) = s;
}

// ---- online softmax over tokens of k[:,c], dotted with v[:,c] (x4 chans) --
__global__ __launch_bounds__(256)
void kctx_kernel(const float* __restrict__ qkv, float* __restrict__ ctx)
{
    const int C4 = DIM / 4;
    const int idx = blockIdx.x * 256 + threadIdx.x;
    if (idx >= BQ * C4) return;
    const int c = (idx % C4) * 4;
    const int b = idx / C4;
    const float* base = qkv + (size_t)b * NTOK * QKVW;

    float m[4], s[4], a[4];
    #pragma unroll
    for (int j = 0; j < 4; ++j) { m[j] = -3.0e38f; s[j] = 0.0f; a[j] = 0.0f; }

    for (int n = 0; n < NTOK; ++n) {
        const float4 kv4 = *(const float4*)(base + (size_t)n * QKVW + DIM + c);
        const float4 vv4 = *(const float4*)(base + (size_t)n * QKVW + 2 * DIM + c);
        const float kv[4] = {kv4.x, kv4.y, kv4.z, kv4.w};
        const float vv[4] = {vv4.x, vv4.y, vv4.z, vv4.w};
        #pragma unroll
        for (int j = 0; j < 4; ++j) {
            const float nm = fmaxf(m[j], kv[j]);
            const float e0 = __expf(m[j] - nm);
            const float e1 = __expf(kv[j] - nm);
            s[j] = s[j] * e0 + e1;
            a[j] = a[j] * e0 + e1 * vv[j];
            m[j] = nm;
        }
    }
    *(float4*)(ctx + (size_t)b * DIM + c) =
        make_float4(a[0] / s[0], a[1] / s[1], a[2] / s[2], a[3] / s[3]);
}

// -------------------- cls-row copy/scale helper (float4) -------------------
__global__ __launch_bounds__(256)
void cls_row(const float* __restrict__ src, int sstride, int soff, float scale,
             float* __restrict__ dst)
{
    const int C4 = DIM / 4;
    const int idx = blockIdx.x * 256 + threadIdx.x;
    if (idx >= BQ * C4) return;
    const int c = (idx % C4) * 4;
    const int b = idx / C4;
    const float4 v = *(const float4*)(src + (size_t)b * NTOK * sstride + soff + c);
    *(float4*)(dst + (size_t)b * NTOK * DIM + c) =
        make_float4(scale * v.x, scale * v.y, scale * v.z, scale * v.w);
}

// ---- epilogue: out_pre = q*(v_final+ctx); attn = q*ctx (float4) -----------
__global__ __launch_bounds__(256)
void epilogue_kernel(const float* __restrict__ qkv, const float* __restrict__ vfin,
                     const float* __restrict__ ctx, float* __restrict__ opre,
                     float* __restrict__ attn)
{
    const int C4 = DIM / 4;
    const size_t idx   = (size_t)blockIdx.x * 256 + threadIdx.x;
    const size_t total = (size_t)BQ * NTOK * C4;
    if (idx >= total) return;
    const int c = (int)(idx % C4) * 4;
    const size_t t = idx / C4;          // b*197 + n
    const int n = (int)(t % NTOK);
    const int b = (int)(t / NTOK);

    const float4 q  = *(const float4*)(qkv + t * QKVW + c);
    const float4 cx = *(const float4*)(ctx + (size_t)b * DIM + c);
    const float4 vf = *(const float4*)(vfin + t * DIM + c);

    const float4 av = make_float4(q.x * cx.x, q.y * cx.y, q.z * cx.z, q.w * cx.w);
    *(float4*)(opre + t * DIM + c) =
        make_float4(q.x * vf.x + av.x, q.y * vf.y + av.y,
                    q.z * vf.z + av.z, q.w * vf.w + av.w);

    const int h  = c >> 6;
    const int jj = c & 63;              // 4-aligned within the 64-wide head
    *(float4*)(attn + (((size_t)b * 12 + h) * NTOK + n) * 64 + jj) = av;
}

// ---------------------------------------------------------------------------
extern "C" void kernel_launch(void* const* d_in, const int* in_sizes, int n_in,
                              void* d_out, int out_size, void* d_ws, size_t ws_size,
                              hipStream_t stream)
{
    const float* x     = (const float*)d_in[0];
    const float* dw1_w = (const float*)d_in[1];
    const float* dw1_b = (const float*)d_in[2];
    const float* pw1_w = (const float*)d_in[3];
    const float* pw1_b = (const float*)d_in[4];
    const float* w_qkv = (const float*)d_in[5];
    const float* dw2_w = (const float*)d_in[6];
    const float* dw2_b = (const float*)d_in[7];
    const float* pw2_w = (const float*)d_in[8];
    const float* pw2_b = (const float*)d_in[9];
    const float* w_out = (const float*)d_in[10];
    const float* b_out = (const float*)d_in[11];

    float* out_main = (float*)d_out;                                 // (B,N,768)
    float* out_attn = out_main + (size_t)BQ * NTOK * DIM;            // (B,12,197,64)

    char* ws = (char*)d_ws;
    // bufA/bufB hold up to 12800 rows (50*256) so GEMM A loads need no guard.
    float*    bufA  = (float*)(ws);                                  // 40 MB
    float*    bufB  = (float*)(ws + ((size_t)40  << 20));            // 40 MB
    float*    qkvb  = (float*)(ws + ((size_t)80  << 20));            // 120 MB
    float*    ctxb  = (float*)(ws + ((size_t)200 << 20));            // 192 KB
    _Float16* whPW1 = (_Float16*)(ws + ((size_t)201 << 20));         // 1.2 MB
    _Float16* whPW2 = (_Float16*)(ws + ((size_t)203 << 20));         // 1.2 MB
    _Float16* whQKV = (_Float16*)(ws + ((size_t)205 << 20));         // 3.6 MB
    _Float16* whOUT = (_Float16*)(ws + ((size_t)209 << 20));         // 1.2 MB

    const int M_IMG = BQ * NIMG;   // 12544 = 49*256
    const int M_ALL = BQ * NTOK;   // 12608 -> 50 blocks, padded reads

    const int C4 = DIM / 4;
    const int ew_grid   = (int)(((size_t)BQ * NIMG * C4 + 255) / 256);
    const int full_grid = (int)(((size_t)BQ * NTOK * C4 + 255) / 256);
    const int cls_grid  = (BQ * C4 + 255) / 256;

    // ---- weight prep (per launch; deterministic) ----
    convert_f16<<<(DIM * DIM / 4 + 255) / 256, 256, 0, stream>>>(pw1_w, whPW1, DIM * DIM / 4);
    convert_f16<<<(DIM * DIM / 4 + 255) / 256, 256, 0, stream>>>(pw2_w, whPW2, DIM * DIM / 4);
    transpose_f16<<<dim3(DIM / 32, QKVW / 32), 256, 0, stream>>>(w_qkv, DIM, QKVW, whQKV);
    transpose_f16<<<dim3(DIM / 32, DIM / 32), 256, 0, stream>>>(w_out, DIM, DIM, whOUT);

    // 1) depthwise conv 1 on x image tokens -> bufA
    dwconv3x3<<<ew_grid, 256, 0, stream>>>(x, DIM, 0, dw1_w, dw1_b, bufA);

    // 2) pointwise conv 1 + bias + residual(x) -> x_aug image rows (bufB)
    gemm_wmma_f16<<<dim3(M_IMG / BM, DIM / BN), 256, 0, stream>>>(
        bufA, DIM, M_IMG, DIM, whPW1, pw1_b, x, DIM, bufB, DIM, /*rowmap=*/1);

    // 3) x_aug cls rows = 2 * x_cls
    cls_row<<<cls_grid, 256, 0, stream>>>(x, DIM, 0, 2.0f, bufB);

    // 4) qkv GEMM: x_aug @ w_qkv -> qkvb
    gemm_wmma_f16<<<dim3((M_ALL + BM - 1) / BM, QKVW / BN), 256, 0, stream>>>(
        bufB, DIM, M_ALL, DIM, whQKV, nullptr, nullptr, 0, qkvb, QKVW, /*rowmap=*/0);

    // 5) softmax(k) . v per channel -> ctx
    kctx_kernel<<<cls_grid, 256, 0, stream>>>(qkvb, ctxb);

    // 6) depthwise conv 2 on v image tokens -> bufA
    dwconv3x3<<<ew_grid, 256, 0, stream>>>(qkvb, QKVW, 2 * DIM, dw2_w, dw2_b, bufA);

    // 7) pointwise conv 2 + bias -> v_final image rows (bufB)
    gemm_wmma_f16<<<dim3(M_IMG / BM, DIM / BN), 256, 0, stream>>>(
        bufA, DIM, M_IMG, DIM, whPW2, pw2_b, nullptr, 0, bufB, DIM, /*rowmap=*/1);

    // 8) v_final cls rows = v_cls
    cls_row<<<cls_grid, 256, 0, stream>>>(qkvb, QKVW, 2 * DIM, 1.0f, bufB);

    // 9) epilogue: out_pre (bufA) and attn output
    epilogue_kernel<<<full_grid, 256, 0, stream>>>(qkvb, bufB, ctxb, bufA, out_attn);

    // 10) output projection: out_pre @ w_out + b_out -> out_main
    gemm_wmma_f16<<<dim3((M_ALL + BM - 1) / BM, DIM / BN), 256, 0, stream>>>(
        bufA, DIM, M_ALL, DIM, whOUT, b_out, nullptr, 0, out_main, DIM, /*rowmap=*/0);
}
